// SelfAttentiom_10857677324557
// MI455X (gfx1250) — matrix operand order repaired and checked by reading, not compiled, not run on hardware
//
#include <hip/hip_runtime.h>

typedef __bf16 bf16;
typedef __attribute__((ext_vector_type(16))) __bf16 v16bf;
typedef __attribute__((ext_vector_type(8)))  __bf16 v8bf;
typedef __attribute__((ext_vector_type(4)))  __bf16 v4bf;
typedef __attribute__((ext_vector_type(8)))  float  v8f;
typedef __attribute__((ext_vector_type(4)))  float  v4f;

#define B_   4
#define S_   2048
#define D_   768
#define H_   12
#define DH_  64
#define M_   (B_ * S_)   /* 8192 */
#define N3_  (3 * D_)    /* 2304 */

// gfx1250 async global->LDS path (ASYNCcnt-tracked, bypasses VGPRs).
#if defined(__gfx1250__) && __has_builtin(__builtin_amdgcn_global_load_async_to_lds_b128) && __has_builtin(__builtin_amdgcn_s_wait_asynccnt)
#define USE_ASYNC 1
#else
#define USE_ASYNC 0
#endif

// Builtin wants: (int4 addrspace(1)* src, int4 addrspace(3)* dst, imm off, imm cpol)
typedef int v4i __attribute__((vector_size(16)));
typedef __attribute__((address_space(1))) v4i* gv4_ptr;
typedef __attribute__((address_space(3))) v4i* lv4_ptr;

static __device__ __forceinline__ void async_b128(const bf16* gsrc, bf16* ldst) {
#if USE_ASYNC
  __builtin_amdgcn_global_load_async_to_lds_b128((gv4_ptr)(v4i*)gsrc, (lv4_ptr)(v4i*)ldst, 0, 0);
#else
  *(v8bf*)ldst = *(const v8bf*)gsrc;
#endif
}

// Build a 16-element bf16 WMMA fragment from two 16-byte LDS/global chunks.
static __device__ __forceinline__ v16bf frag16(const bf16* p0, const bf16* p1) {
  union { v16bf v; v8bf h[2]; } u;
  u.h[0] = *(const v8bf*)p0;
  u.h[1] = *(const v8bf*)p1;
  return u.v;
}

static __device__ __forceinline__ v8f wmma_bf16(v16bf a, v16bf b, v8f c) {
  return __builtin_amdgcn_wmma_f32_16x16x32_bf16(
      /*neg_a=*/false, a, /*neg_b=*/false, b,
      /*c_mod=*/(short)0, c, /*reuse_a=*/false, /*reuse_b=*/false);
}

// ---------------------------------------------------------------------------
// Kernel 1: fused QKV projection.  qkv[m,e] = sum_k X[m,k] * Win[e,k]
// Writes Q,K as bf16 [B,H,S,64] and V transposed as bf16 [B,H,64,S].
// 128x128 block tile, 8 waves (2x4), each wave 64x32 = 8 WMMA tiles.
// ---------------------------------------------------------------------------
__global__ __launch_bounds__(256) void qkv_proj_kernel(
    const float* __restrict__ X, const float* __restrict__ W,
    bf16* __restrict__ qw, bf16* __restrict__ kw, bf16* __restrict__ vT) {
  __shared__ bf16 As[128 * 32];   // [m][k]
  __shared__ bf16 Bs[128 * 32];   // [n][k]

  const int tid  = threadIdx.x;
  const int Mb   = blockIdx.x * 128;
  const int Nb   = blockIdx.y * 128;
  const int w    = tid >> 5, lane = tid & 31;
  const int wr   = w >> 2,  wc   = w & 3;
  const int half = lane >> 4, l16 = lane & 15;

  v8f acc[4][2] = {};

  for (int k0 = 0; k0 < D_; k0 += 32) {
    __syncthreads();
    // Cooperative load + fp32->bf16 convert: 128x32 of A and B each.
#pragma unroll
    for (int it = 0; it < 4; ++it) {
      int idx = tid * 4 + it;            // 0..1023
      int row = idx >> 3;
      int kq  = (idx & 7) * 4;
      v4f fa = *(const v4f*)(X + (size_t)(Mb + row) * D_ + k0 + kq);
      v4f fb = *(const v4f*)(W + (size_t)(Nb + row) * D_ + k0 + kq);
      v4bf ca, cb;
#pragma unroll
      for (int e = 0; e < 4; ++e) { ca[e] = (bf16)fa[e]; cb[e] = (bf16)fb[e]; }
      *(v4bf*)&As[row * 32 + kq] = ca;
      *(v4bf*)&Bs[row * 32 + kq] = cb;
    }
    __syncthreads();

    v16bf bfr[2];
#pragma unroll
    for (int j = 0; j < 2; ++j) {
      const bf16* bp = &Bs[(wc * 32 + j * 16 + l16) * 32 + half * 16];
      bfr[j] = frag16(bp, bp + 8);       // B: 16 contiguous K per lane
    }
#pragma unroll
    for (int i = 0; i < 4; ++i) {
      const bf16* ap = &As[(wr * 64 + i * 16 + l16) * 32 + half * 16];
      v16bf afr = frag16(ap, ap + 8);    // A: 16 contiguous K per lane
#pragma unroll
      for (int j = 0; j < 2; ++j) acc[i][j] = wmma_bf16(afr, bfr[j], acc[i][j]);
    }
  }

  const int seg = Nb / D_;  // 0=Q 1=K 2=V, uniform per block (768 % 128 == 0)
#pragma unroll
  for (int i = 0; i < 4; ++i) {
#pragma unroll
    for (int j = 0; j < 2; ++j) {
      int col  = Nb + wc * 32 + j * 16 + l16;
      int e    = col - seg * D_;
      int h    = e >> 6, dcol = e & 63;
#pragma unroll
      for (int r = 0; r < 8; ++r) {
        int m = Mb + wr * 64 + i * 16 + r + 8 * half;
        int b = m >> 11, s = m & (S_ - 1);
        bf16 val = (bf16)acc[i][j][r];
        if (seg == 0)
          qw[(((size_t)(b * H_ + h)) * S_ + s) * DH_ + dcol] = val;
        else if (seg == 1)
          kw[(((size_t)(b * H_ + h)) * S_ + s) * DH_ + dcol] = val;
        else
          vT[(((size_t)(b * H_ + h)) * DH_ + dcol) * S_ + s] = val;  // V transposed
      }
    }
  }
}

// ---------------------------------------------------------------------------
// Kernel 2: flash attention with double-buffered async global->LDS K/V tiles.
// One WG = 128 query rows of one (b,h); 8 waves, 16 q-rows per wave.
// kv tiles of 64 with online softmax; S x S scores never materialized.
// ---------------------------------------------------------------------------
__global__ __launch_bounds__(256) void flash_attn_kernel(
    const bf16* __restrict__ qw, const bf16* __restrict__ kw,
    const bf16* __restrict__ vT, bf16* __restrict__ attn) {
  __shared__ bf16 Ks[2][64 * 64];     // [t][d], double buffered
  __shared__ bf16 Vs[2][64 * 64];     // [d][t] (V pre-transposed in ws)
  __shared__ bf16 Ps[8 * 16 * 64];    // per-wave P staging (C-layout -> A-layout)

  const int tid = threadIdx.x;
  const int qt  = blockIdx.x & 15;
  const int bh  = blockIdx.x >> 4;
  const int b   = bh / H_, h = bh - b * H_;

  const bf16* Qb = qw + (size_t)bh * S_ * DH_;
  const bf16* Kb = kw + (size_t)bh * S_ * DH_;
  const bf16* Vb = vT + (size_t)bh * DH_ * S_;

  const int w = tid >> 5, lane = tid & 31;
  const int half = lane >> 4, l16 = lane & 15;
  const int q0 = qt * 128;
  const int qrow = q0 + w * 16 + l16;

  // Q fragments live in registers for the whole kernel (16x64 per wave).
  v16bf qfr[2];
#pragma unroll
  for (int f = 0; f < 2; ++f) {
    const bf16* p = Qb + (size_t)qrow * DH_ + f * 32 + half * 16;
    qfr[f] = frag16(p, p + 8);
  }

  v8f O[4] = {};
  float mrow[8], lrow[8];
#pragma unroll
  for (int r = 0; r < 8; ++r) { mrow[r] = -__builtin_inff(); lrow[r] = 0.f; }

  const float scale = 0.125f;   // 1/sqrt(64)
  bf16* Pw = &Ps[w * (16 * 64)];

  // Per-thread slice of each 64x64 tile: 4 x 16B (K row chunk x2, V row chunk x2).
  const int trow = tid >> 2;            // 0..63
  const int tcq  = (tid & 3) * 16;      // 0,16,32,48
  const bf16* kg0 = Kb + (size_t)trow * DH_ + tcq;  // + t0*DH_
  const bf16* vg0 = Vb + (size_t)trow * S_  + tcq;  // + t0

  auto load_tile = [&](int buf, int t0) {
    const bf16* ks = kg0 + (size_t)t0 * DH_;
    const bf16* vs = vg0 + t0;
    bf16* kd = &Ks[buf][trow * 64 + tcq];
    bf16* vd = &Vs[buf][trow * 64 + tcq];
    async_b128(ks,     kd);
    async_b128(ks + 8, kd + 8);
    async_b128(vs,     vd);
    async_b128(vs + 8, vd + 8);
  };

  load_tile(0, 0);   // prologue: tile 0 in flight

  for (int it = 0; it < S_ / 64; ++it) {
    const int cur = it & 1;
#if USE_ASYNC
    __builtin_amdgcn_s_wait_asynccnt(0);   // own tile-it loads landed in LDS
#endif
    __syncthreads();                        // all waves: loads landed, prev reads done
    if (it + 1 < S_ / 64) load_tile(1 - cur, (it + 1) * 64);  // overlap with compute

    const bf16* Kc = &Ks[cur][0];
    const bf16* Vc = &Vs[cur][0];

    // S_tile = Q (16x64) @ K^T (64x64) : 8 WMMAs
    v8f sacc[4] = {};
#pragma unroll
    for (int f = 0; f < 2; ++f) {
#pragma unroll
      for (int j = 0; j < 4; ++j) {
        const bf16* kp = &Kc[(j * 16 + l16) * 64 + f * 32 + half * 16];
        v16bf kfr = frag16(kp, kp + 8);
        sacc[j] = wmma_bf16(qfr[f], kfr, sacc[j]);
      }
    }

    // Online softmax per owned row (row = r + 8*half).
#pragma unroll
    for (int r = 0; r < 8; ++r) {
      float s0 = sacc[0][r] * scale;
      float s1 = sacc[1][r] * scale;
      float s2 = sacc[2][r] * scale;
      float s3 = sacc[3][r] * scale;
      float tmax = fmaxf(fmaxf(s0, s1), fmaxf(s2, s3));
#pragma unroll
      for (int off = 8; off >= 1; off >>= 1)
        tmax = fmaxf(tmax, __shfl_xor(tmax, off, 32));
      float mnew  = fmaxf(mrow[r], tmax);
      float alpha = __expf(mrow[r] - mnew);
      float p0 = __expf(s0 - mnew);
      float p1 = __expf(s1 - mnew);
      float p2 = __expf(s2 - mnew);
      float p3 = __expf(s3 - mnew);
      float rs = (p0 + p1) + (p2 + p3);
#pragma unroll
      for (int off = 8; off >= 1; off >>= 1)
        rs += __shfl_xor(rs, off, 32);
      lrow[r] = lrow[r] * alpha + rs;
      mrow[r] = mnew;
      O[0][r] *= alpha; O[1][r] *= alpha; O[2][r] *= alpha; O[3][r] *= alpha;
      int prow = r + 8 * half;
      Pw[prow * 64 +  0 + l16] = (bf16)p0;
      Pw[prow * 64 + 16 + l16] = (bf16)p1;
      Pw[prow * 64 + 32 + l16] = (bf16)p2;
      Pw[prow * 64 + 48 + l16] = (bf16)p3;
    }

    __syncthreads();   // P visible for A-layout reload

    // O += P (16x64) @ V (64x64) : 8 WMMAs
#pragma unroll
    for (int f = 0; f < 2; ++f) {
      const bf16* pp = &Pw[l16 * 64 + f * 32 + half * 16];
      v16bf pfr = frag16(pp, pp + 8);
#pragma unroll
      for (int ot = 0; ot < 4; ++ot) {
        const bf16* vp = &Vc[(ot * 16 + l16) * 64 + f * 32 + half * 16];
        v16bf vfr = frag16(vp, vp + 8);
        O[ot] = wmma_bf16(pfr, vfr, O[ot]);
      }
    }
  }

  // Normalize and store attention output [B,S,D] (heads merged) as bf16.
  float inv[8];
#pragma unroll
  for (int r = 0; r < 8; ++r) inv[r] = 1.0f / lrow[r];
#pragma unroll
  for (int ot = 0; ot < 4; ++ot) {
#pragma unroll
    for (int r = 0; r < 8; ++r) {
      int m   = q0 + w * 16 + r + 8 * half;
      int col = h * DH_ + ot * 16 + l16;
      attn[((size_t)b * S_ + m) * D_ + col] = (bf16)(O[ot][r] * inv[r]);
    }
  }
}

// ---------------------------------------------------------------------------
// Kernel 3: output projection.  out[m,n] = sum_k attn[m,k] * Wout[n,k], f32 out.
// ---------------------------------------------------------------------------
__global__ __launch_bounds__(256) void out_proj_kernel(
    const bf16* __restrict__ A, const float* __restrict__ W,
    float* __restrict__ out) {
  __shared__ bf16 As[128 * 32];
  __shared__ bf16 Bs[128 * 32];

  const int tid  = threadIdx.x;
  const int Mb   = blockIdx.x * 128;
  const int Nb   = blockIdx.y * 128;
  const int w    = tid >> 5, lane = tid & 31;
  const int wr   = w >> 2,  wc   = w & 3;
  const int half = lane >> 4, l16 = lane & 15;

  v8f acc[4][2] = {};

  for (int k0 = 0; k0 < D_; k0 += 32) {
    __syncthreads();
    {
      int arow = tid >> 1;
      int acq  = (tid & 1) * 16;
      const bf16* ap = A + (size_t)(Mb + arow) * D_ + k0 + acq;
      *(v8bf*)&As[arow * 32 + acq]     = *(const v8bf*)ap;
      *(v8bf*)&As[arow * 32 + acq + 8] = *(const v8bf*)(ap + 8);
    }
#pragma unroll
    for (int it = 0; it < 4; ++it) {
      int idx = tid * 4 + it;
      int row = idx >> 3;
      int kq  = (idx & 7) * 4;
      v4f fb = *(const v4f*)(W + (size_t)(Nb + row) * D_ + k0 + kq);
      v4bf cb;
#pragma unroll
      for (int e = 0; e < 4; ++e) cb[e] = (bf16)fb[e];
      *(v4bf*)&Bs[row * 32 + kq] = cb;
    }
    __syncthreads();

    v16bf bfr[2];
#pragma unroll
    for (int j = 0; j < 2; ++j) {
      const bf16* bp = &Bs[(wc * 32 + j * 16 + l16) * 32 + half * 16];
      bfr[j] = frag16(bp, bp + 8);
    }
#pragma unroll
    for (int i = 0; i < 4; ++i) {
      const bf16* ap = &As[(wr * 64 + i * 16 + l16) * 32 + half * 16];
      v16bf afr = frag16(ap, ap + 8);
#pragma unroll
      for (int j = 0; j < 2; ++j) acc[i][j] = wmma_bf16(afr, bfr[j], acc[i][j]);
    }
  }

#pragma unroll
  for (int i = 0; i < 4; ++i) {
#pragma unroll
    for (int j = 0; j < 2; ++j) {
      int col = Nb + wc * 32 + j * 16 + l16;
#pragma unroll
      for (int r = 0; r < 8; ++r) {
        int m = Mb + wr * 64 + i * 16 + r + 8 * half;
        out[(size_t)m * D_ + col] = acc[i][j][r];
      }
    }
  }
}

extern "C" void kernel_launch(void* const* d_in, const int* in_sizes, int n_in,
                              void* d_out, int out_size, void* d_ws, size_t ws_size,
                              hipStream_t stream) {
  (void)in_sizes; (void)n_in; (void)out_size; (void)ws_size;
  const float* x  = (const float*)d_in[0];   // [4, 2048, 768]
  const float* wi = (const float*)d_in[1];   // [2304, 768]
  const float* wo = (const float*)d_in[2];   // [768, 768]
  float* out = (float*)d_out;                // [4, 2048, 768] f32

  // Workspace partitions (bf16): Q, K, V^T, attn-out = 4 x 12 MB = 50 MB.
  const size_t per = (size_t)B_ * H_ * S_ * DH_;   // 6,291,456 elements
  bf16* qw   = (bf16*)d_ws;
  bf16* kw   = qw + per;
  bf16* vT   = kw + per;
  bf16* attn = vT + per;

  qkv_proj_kernel<<<dim3(M_ / 128, N3_ / 128), 256, 0, stream>>>(x, wi, qw, kw, vT);
  flash_attn_kernel<<<dim3(B_ * H_ * (S_ / 128)), 256, 0, stream>>>(qw, kw, vT, attn);
  out_proj_kernel<<<dim3(M_ / 128, D_ / 128), 256, 0, stream>>>(attn, wo, out);
}
